// FlexSelfAttention_10222022164658
// MI455X (gfx1250) — matrix operand order repaired and checked by reading, not compiled, not run on hardware
//
#include <hip/hip_runtime.h>
#include <hip/hip_bf16.h>

// ---------------------------------------------------------------------------
// FlexSelfAttention for MI455X (gfx1250, wave32, WMMA, async-to-LDS staging)
// B=1, S=2048, HQ=16, HKV=4, D=128, HIDDEN=2048
// Pipeline:
//   gemm<f16 out> Q,K   gemm<f16 transposed out> V
//   rmsnorm+rope (q,k in f16)
//   flash attention (f16 tiles staged via global_load_async_to_lds_b128)
//   gemm<f32 out> O-projection
// ---------------------------------------------------------------------------

typedef __attribute__((ext_vector_type(16))) _Float16 v16h;
typedef __attribute__((ext_vector_type(8)))  _Float16 v8h;
typedef __attribute__((ext_vector_type(4)))  _Float16 v4h;
typedef __attribute__((ext_vector_type(8)))  float    v8f;
typedef __attribute__((ext_vector_type(4)))  float    v4f;
typedef __attribute__((ext_vector_type(4)))  int      v4i;

union F16x16 {
    v16h v;
    v8h  h[2];
    _Float16 e[16];
};

constexpr int S_LEN  = 2048;
constexpr int HQ_N   = 16;
constexpr int HKV_N  = 4;
constexpr int D_DIM  = 128;
constexpr int HIDDEN = HQ_N * D_DIM;      // 2048
constexpr int KVDIM  = HKV_N * D_DIM;     // 512
constexpr float RMS_EPS   = 1.1920929e-07f;
constexpr float ROPE_BASE = 10000.0f;

#define LSTR 40   // LDS row stride in halfs (80 bytes, 16B aligned)

// ---------------------------------------------------------------------------
// Async global->LDS copy (16B per lane). When available this lowers to
// global_load_async_to_lds_b128 (tracked by ASYNCcnt -> s_wait_asynccnt).
// Builtin signature (from hipcc diagnostic): (v4i AS1* gsrc, v4i AS3* ldst,
// imm offset, imm cpol).
// ---------------------------------------------------------------------------
#if defined(__AMDGCN__) && __has_builtin(__builtin_amdgcn_global_load_async_to_lds_b128)
#define USE_ASYNC 1
#else
#define USE_ASYNC 0
#endif

__device__ __forceinline__ void async_copy_b128(const _Float16* g, _Float16* l)
{
#if USE_ASYNC
    __attribute__((address_space(1))) v4i* gp =
        (__attribute__((address_space(1))) v4i*)(uintptr_t)g;
    __attribute__((address_space(3))) v4i* lp =
        (__attribute__((address_space(3))) v4i*)(unsigned int)(uintptr_t)l;
    __builtin_amdgcn_global_load_async_to_lds_b128(gp, lp, 0, 0);
#else
    *(v8h*)l = *(const v8h*)g;
#endif
}

__device__ __forceinline__ void async_copy_wait()
{
#if USE_ASYNC
    asm volatile("s_wait_asynccnt 0" ::: "memory");
#endif
}

// ---------------------------------------------------------------------------
// GEMM: C[M,N] = A[M,K] @ B[N,K]^T  (A,B f32; f16 WMMA compute, f32 acc)
// MODE 0: C f32 row-major.  MODE 1: C f16 row-major.  MODE 2: C f16 col-major
// (C[col*M+row], i.e. transposed output for V).
// Block: 256 threads = 8 waves. Block tile 128x128, K step 32.
// Wave (w&3)*32 rows x (w>>2)*64 cols -> 8 accumulators (2x4 of 16x16).
// ---------------------------------------------------------------------------
template <int MODE>
__global__ __launch_bounds__(256)
void gemm_xwt_wmma(const float* __restrict__ A, const float* __restrict__ B,
                   void* __restrict__ Cv, int M, int N, int K)
{
    __shared__ __align__(16) _Float16 As[128 * LSTR];
    __shared__ __align__(16) _Float16 Bs[128 * LSTR];

    const int tid  = threadIdx.x;
    const int wave = tid >> 5;
    const int lane = tid & 31;
    const int hi   = lane >> 4;
    const int l16  = lane & 15;
    const int wm   = (wave & 3) * 32;    // wave row offset in block tile
    const int wn   = (wave >> 2) * 64;   // wave col offset in block tile
    const int bm   = blockIdx.y * 128;
    const int bn   = blockIdx.x * 128;

    v8f acc[2][4];
    v8f zero = {};
#pragma unroll
    for (int mi = 0; mi < 2; ++mi)
#pragma unroll
        for (int ni = 0; ni < 4; ++ni) acc[mi][ni] = zero;

    for (int k0 = 0; k0 < K; k0 += 32) {
        // cooperative staging: 128x32 f32 -> f16, vectorized (float4 -> v4h)
#pragma unroll
        for (int i = 0; i < 4; ++i) {
            int e = tid + i * 256;      // float4 index (1024 per tile)
            int r = e >> 3;             // 8 float4s per 32-col row
            int c = (e & 7) * 4;
            v4f a4 = *(const v4f*)&A[(size_t)(bm + r) * K + (k0 + c)];
            v4f b4 = *(const v4f*)&B[(size_t)(bn + r) * K + (k0 + c)];
            v4h ah, bh;
#pragma unroll
            for (int j = 0; j < 4; ++j) {
                ah[j] = (_Float16)a4[j];
                bh[j] = (_Float16)b4[j];
            }
            *(v4h*)&As[r * LSTR + c] = ah;
            *(v4h*)&Bs[r * LSTR + c] = bh;
            if (k0 + 32 < K) {
                __builtin_prefetch(&A[(size_t)(bm + r) * K + (k0 + 32 + c)], 0, 1);
                __builtin_prefetch(&B[(size_t)(bn + r) * K + (k0 + 32 + c)], 0, 1);
            }
        }
        __syncthreads();

        // A fragments (16x32): lane m = l16; K halves split by lane-half
        F16x16 af[2];
#pragma unroll
        for (int mi = 0; mi < 2; ++mi) {
            const _Float16* ap = &As[(wm + mi * 16 + l16) * LSTR];
            af[mi].h[0] = *(const v8h*)(ap + hi * 8);
            af[mi].h[1] = *(const v8h*)(ap + 16 + hi * 8);
        }
        // B fragments (32x16): lane n = l16; K halves split by lane-half (same as A)
#pragma unroll
        for (int ni = 0; ni < 4; ++ni) {
            const _Float16* bp = &Bs[(wn + ni * 16 + l16) * LSTR];
            F16x16 bf;
            bf.h[0] = *(const v8h*)(bp + hi * 8);
            bf.h[1] = *(const v8h*)(bp + 16 + hi * 8);
#pragma unroll
            for (int mi = 0; mi < 2; ++mi)
                acc[mi][ni] = __builtin_amdgcn_wmma_f32_16x16x32_f16(
                    false, af[mi].v, false, bf.v, (short)0, acc[mi][ni], false, false);
        }
        __syncthreads();
    }

    // C/D layout: lanes 0-15 -> M=r, lanes 16-31 -> M=8+r; N = l16
#pragma unroll
    for (int mi = 0; mi < 2; ++mi)
#pragma unroll
        for (int ni = 0; ni < 4; ++ni)
#pragma unroll
            for (int r = 0; r < 8; ++r) {
                int row = bm + wm + mi * 16 + hi * 8 + r;
                int col = bn + wn + ni * 16 + l16;
                float val = acc[mi][ni][r];
                if constexpr (MODE == 0)
                    ((float*)Cv)[(size_t)row * N + col] = val;
                else if constexpr (MODE == 1)
                    ((_Float16*)Cv)[(size_t)row * N + col] = (_Float16)val;
                else
                    ((_Float16*)Cv)[(size_t)col * M + row] = (_Float16)val;
            }
}

// ---------------------------------------------------------------------------
// RMS-norm (per head, D=128) + rotary embedding, in place on f16 q and k.
// grid (S, HQ+HKV), 128 threads. Position == row index (x_pos = arange(S)).
// ---------------------------------------------------------------------------
__global__ __launch_bounds__(128)
void rmsrope_kernel(_Float16* __restrict__ qg, _Float16* __restrict__ kg)
{
    const int s = blockIdx.x;
    const int h = blockIdx.y;
    const int t = threadIdx.x;

    _Float16* ptr = (h < HQ_N)
        ? (qg + (size_t)s * HIDDEN + h * D_DIM)
        : (kg + (size_t)s * KVDIM + (h - HQ_N) * D_DIM);

    float val = (float)ptr[t];
    float ss = val * val;
#pragma unroll
    for (int m = 16; m >= 1; m >>= 1) ss += __shfl_xor(ss, m, 32);

    __shared__ float wsum[4];
    __shared__ float nrm[D_DIM];
    const int wave = t >> 5, lane = t & 31;
    if (lane == 0) wsum[wave] = ss;
    __syncthreads();
    float tot = wsum[0] + wsum[1] + wsum[2] + wsum[3];
    float r = rsqrtf(tot * (1.0f / D_DIM) + RMS_EPS);
    nrm[t] = val * r;
    __syncthreads();

    if (t < D_DIM / 2) {
        float x1 = nrm[t];
        float x2 = nrm[t + D_DIM / 2];
        float inv = __powf(ROPE_BASE, -(float)t / (float)(D_DIM / 2));
        float fr = (float)s * inv;
        float sn, cs;
        __sincosf(fr, &sn, &cs);
        ptr[t]             = (_Float16)( x1 * cs + x2 * sn);
        ptr[t + D_DIM / 2] = (_Float16)(-x1 * sn + x2 * cs);
    }
}

// ---------------------------------------------------------------------------
// Causal flash attention (GQA g=4). grid (S/64, HQ), 128 threads = 4 waves.
// Each wave owns 16 q rows; kv in 32-column tiles. Q/K/V tiles staged from
// f16 global via async b128 copies (V comes pre-transposed per head).
// ---------------------------------------------------------------------------
__global__ __launch_bounds__(128)
void flash_attn_kernel(const _Float16* __restrict__ qg, const _Float16* __restrict__ kg,
                       const _Float16* __restrict__ vtg, float* __restrict__ og)
{
    constexpr int QT = 64;            // block q rows
    constexpr int KT = 32;            // kv tile columns
    constexpr int DSTR = D_DIM + 8;   // 136 halfs, 272B rows (16B aligned)

    __shared__ __align__(16) _Float16 Qs[QT * DSTR];       // q tile [qrow][d]
    __shared__ __align__(16) _Float16 Ks[KT * DSTR];       // k tile [kvrow][d]
    __shared__ __align__(16) _Float16 Vt[D_DIM * LSTR];    // v tile transposed [d][kvrow]
    __shared__ __align__(16) _Float16 Ps[4 * 16 * LSTR];   // per-wave P tile [16][32]

    const int tid  = threadIdx.x;
    const int wave = tid >> 5;
    const int lane = tid & 31;
    const int hi   = lane >> 4;
    const int l16  = lane & 15;

    const int h  = blockIdx.y;
    const int hk = h / (HQ_N / HKV_N);
    const int b0 = blockIdx.x * QT;
    const int qb = b0 + wave * 16;

    // stage Q tile (64 x 128 halfs) -- 1024 b128 chunks, 8 per thread
#pragma unroll
    for (int i = 0; i < 8; ++i) {
        int e = tid + i * 128;
        int r = e >> 4;                 // 16 chunks per 128-half row
        int c = (e & 15) * 8;
        async_copy_b128(&qg[(size_t)(b0 + r) * HIDDEN + h * D_DIM + c],
                        &Qs[r * DSTR + c]);
    }
    async_copy_wait();
    __syncthreads();

    // persistent Q fragments: 4 chunks of K-dim 32
    F16x16 qf[4];
    const _Float16* qp = &Qs[(wave * 16 + l16) * DSTR];
#pragma unroll
    for (int dc = 0; dc < 4; ++dc) {
        qf[dc].h[0] = *(const v8h*)(qp + dc * 32 + hi * 8);
        qf[dc].h[1] = *(const v8h*)(qp + dc * 32 + 16 + hi * 8);
    }

    v8f zero = {};
    v8f oacc[8];
#pragma unroll
    for (int ch = 0; ch < 8; ++ch) oacc[ch] = zero;
    float mrow[8], lrow[8];
#pragma unroll
    for (int r = 0; r < 8; ++r) { mrow[r] = -__builtin_inff(); lrow[r] = 0.0f; }

    const float scale = 0.08838834764831845f;  // 1/sqrt(128)
    const int kend = b0 + QT;

    for (int kt = 0; kt < kend; kt += KT) {
        __syncthreads();   // previous iteration's consumers done before overwrite
        // stage K tile (32 x 128 halfs): 512 chunks, 4 per thread
#pragma unroll
        for (int i = 0; i < 4; ++i) {
            int e = tid + i * 128;
            int r = e >> 4;
            int c = (e & 15) * 8;
            async_copy_b128(&kg[(size_t)(kt + r) * KVDIM + hk * D_DIM + c],
                            &Ks[r * DSTR + c]);
        }
        // stage V tile transposed [d][kv] from pre-transposed global [hk*D+d][s]
#pragma unroll
        for (int i = 0; i < 4; ++i) {
            int e = tid + i * 128;
            int r = e >> 2;                 // d row (0..127), 4 chunks per 32-half row
            int c = (e & 3) * 8;
            async_copy_b128(&vtg[(size_t)(hk * D_DIM + r) * S_LEN + kt + c],
                            &Vt[r * LSTR + c]);
        }
        async_copy_wait();
        __syncthreads();

        if (kt <= qb + 15) {   // wave has unmasked work in this tile
            // --- scores: S = Q(16x128) @ K^T(128x32) -> two 16x16 accumulators
            v8f s0 = zero, s1 = zero;
#pragma unroll
            for (int dc = 0; dc < 4; ++dc) {
                const _Float16* kp0 = &Ks[l16 * DSTR + dc * 32];
                F16x16 bk0;
                bk0.h[0] = *(const v8h*)(kp0 + hi * 8);
                bk0.h[1] = *(const v8h*)(kp0 + 16 + hi * 8);
                s0 = __builtin_amdgcn_wmma_f32_16x16x32_f16(false, qf[dc].v, false, bk0.v,
                                                            (short)0, s0, false, false);
                const _Float16* kp1 = &Ks[(16 + l16) * DSTR + dc * 32];
                F16x16 bk1;
                bk1.h[0] = *(const v8h*)(kp1 + hi * 8);
                bk1.h[1] = *(const v8h*)(kp1 + 16 + hi * 8);
                s1 = __builtin_amdgcn_wmma_f32_16x16x32_f16(false, qf[dc].v, false, bk1.v,
                                                            (short)0, s1, false, false);
            }

            // --- online softmax; C layout: lane = N col, vgpr r = M row (+8 for hi)
            _Float16* pp = &Ps[wave * 16 * LSTR];
#pragma unroll
            for (int r = 0; r < 8; ++r) {
                const int M = hi * 8 + r;
                const int qrow = qb + M;
                float a0 = s0[r] * scale;
                float a1 = s1[r] * scale;
                if (kt + l16 > qrow)      a0 = -__builtin_inff();
                if (kt + 16 + l16 > qrow) a1 = -__builtin_inff();
                float mx = fmaxf(a0, a1);
                mx = fmaxf(mx, __shfl_xor(mx, 1, 32));
                mx = fmaxf(mx, __shfl_xor(mx, 2, 32));
                mx = fmaxf(mx, __shfl_xor(mx, 4, 32));
                mx = fmaxf(mx, __shfl_xor(mx, 8, 32));
                float mn = fmaxf(mrow[r], mx);
                float corr = __expf(mrow[r] - mn);
                float p0 = __expf(a0 - mn);
                float p1 = __expf(a1 - mn);
                float rs = p0 + p1;
                rs += __shfl_xor(rs, 1, 32);
                rs += __shfl_xor(rs, 2, 32);
                rs += __shfl_xor(rs, 4, 32);
                rs += __shfl_xor(rs, 8, 32);
                lrow[r] = lrow[r] * corr + rs;
                mrow[r] = mn;
#pragma unroll
                for (int ch = 0; ch < 8; ++ch) oacc[ch][r] *= corr;
                pp[M * LSTR + l16]      = (_Float16)p0;
                pp[M * LSTR + 16 + l16] = (_Float16)p1;
            }
            // cross-lane exchange through LDS within this wave
            asm volatile("s_wait_dscnt 0" ::: "memory");

            // P fragment (A-matrix layout, 16x32)
            const _Float16* pr = &Ps[wave * 16 * LSTR + l16 * LSTR];
            F16x16 pf;
            pf.h[0] = *(const v8h*)(pr + hi * 8);
            pf.h[1] = *(const v8h*)(pr + 16 + hi * 8);

            // --- O += P(16x32) @ V(32x128); V B-frags contiguous (transposed tile)
#pragma unroll
            for (int ch = 0; ch < 8; ++ch) {
                const _Float16* vp = &Vt[(ch * 16 + l16) * LSTR];
                F16x16 bv;
                bv.h[0] = *(const v8h*)(vp + hi * 8);
                bv.h[1] = *(const v8h*)(vp + 16 + hi * 8);
                oacc[ch] = __builtin_amdgcn_wmma_f32_16x16x32_f16(false, pf.v, false, bv.v,
                                                                  (short)0, oacc[ch], false, false);
            }
        }
    }

    // normalize and write out [s][h*D + d] as f32
#pragma unroll
    for (int r = 0; r < 8; ++r) {
        const int row = qb + hi * 8 + r;
        const float invl = 1.0f / lrow[r];
#pragma unroll
        for (int ch = 0; ch < 8; ++ch)
            og[(size_t)row * HIDDEN + h * D_DIM + ch * 16 + l16] = oacc[ch][r] * invl;
    }
}

// ---------------------------------------------------------------------------
// Launch.  inputs: 0=x, 1=x_pos (positions == row index), 2=Wq, 3=Wk, 4=Wv, 5=Wo
// ws: q f16 (8MB) | k f16 (2MB) | v^T f16 (2MB) | attn_out f32 (16MB) = 28MB
// ---------------------------------------------------------------------------
extern "C" void kernel_launch(void* const* d_in, const int* in_sizes, int n_in,
                              void* d_out, int out_size, void* d_ws, size_t ws_size,
                              hipStream_t stream)
{
    (void)in_sizes; (void)n_in; (void)out_size; (void)ws_size;

    const float* x  = (const float*)d_in[0];
    const float* Wq = (const float*)d_in[2];
    const float* Wk = (const float*)d_in[3];
    const float* Wv = (const float*)d_in[4];
    const float* Wo = (const float*)d_in[5];
    float* out = (float*)d_out;

    char* ws = (char*)d_ws;
    const size_t QB = (size_t)S_LEN * HIDDEN * sizeof(_Float16);  // 8MB
    const size_t KB = (size_t)S_LEN * KVDIM * sizeof(_Float16);   // 2MB
    _Float16* q  = (_Float16*)(ws);
    _Float16* k  = (_Float16*)(ws + QB);
    _Float16* vt = (_Float16*)(ws + QB + KB);
    float*    ao = (float*)(ws + QB + 2 * KB);

    dim3 blk(256);
    // QKV projections (q,k: f16 row-major; v: f16 transposed per-head [n][s])
    gemm_xwt_wmma<1><<<dim3(HIDDEN / 128, S_LEN / 128), blk, 0, stream>>>(x, Wq, q,  S_LEN, HIDDEN, HIDDEN);
    gemm_xwt_wmma<1><<<dim3(KVDIM / 128,  S_LEN / 128), blk, 0, stream>>>(x, Wk, k,  S_LEN, KVDIM, HIDDEN);
    gemm_xwt_wmma<2><<<dim3(KVDIM / 128,  S_LEN / 128), blk, 0, stream>>>(x, Wv, vt, S_LEN, KVDIM, HIDDEN);
    // RMS norm + rotary on q, k (in place, f16)
    rmsrope_kernel<<<dim3(S_LEN, HQ_N + HKV_N), dim3(128), 0, stream>>>(q, k);
    // causal GQA flash attention
    flash_attn_kernel<<<dim3(S_LEN / 64, HQ_N), dim3(128), 0, stream>>>(q, k, vt, ao);
    // output projection (f32 out)
    gemm_xwt_wmma<0><<<dim3(HIDDEN / 128, S_LEN / 128), blk, 0, stream>>>(ao, Wo, out, S_LEN, HIDDEN, HIDDEN);
}
